// RasterToGraphMWT_60894046322889
// MI455X (gfx1250) — matrix operand order repaired and checked by reading, not compiled, not run on hardware
//
#include <hip/hip_runtime.h>
#include <hip/hip_bf16.h>
#include <math.h>

// ---------------------------------------------------------------------------
// RasterToGraphMWT for MI455X (gfx1250, wave32, WMMA)
// Dominant cost: [N^2,256] x [256,128] pairwise edge-MLP GEMM -> WMMA f16.
// ---------------------------------------------------------------------------

typedef __attribute__((ext_vector_type(16))) _Float16 v16h;
typedef __attribute__((ext_vector_type(8)))  float    v8f;

#define NN   576
#define TT   1000
#define EE   64
#define GG   64
#define HH   256
#define SIDE 24

// ---------------- K0: convert e_w2 [128,256] f32 -> transposed f16 [256,128]
__global__ void k_cvt_w2(const float* __restrict__ w2, _Float16* __restrict__ w2t) {
    int i = blockIdx.x * blockDim.x + threadIdx.x;   // 128*256
    if (i >= 128 * 256) return;
    int n = i >> 8;          // row of e_w2 (output channel, 0..127)
    int k = i & 255;         // col (K, 0..255)
    w2t[k * 128 + n] = (_Float16)w2[i];
}

// ---------------- K1: input projection (per-neuron 2-layer MLP over time)
__global__ void k_inproj(const float* __restrict__ raster,
                         const float* __restrict__ w1, const float* __restrict__ b1,
                         const float* __restrict__ w2, const float* __restrict__ b2,
                         float* __restrict__ x) {
    __shared__ float h[128];
    int n = blockIdx.x;          // 576 neurons
    int t = threadIdx.x;         // 128 threads
    const float* r = raster + n * TT;
    const float* w = w1 + t * TT;
    float acc = b1[t];
    for (int k = 0; k < TT; ++k) acc += r[k] * w[k];
    h[t] = fmaxf(acc, 0.f);
    __syncthreads();
    if (t < EE) {
        float a = b2[t];
        const float* wr = w2 + t * 128;
        for (int k = 0; k < 128; ++k) a += h[k] * wr[k];
        x[n * EE + t] = a;       // x: [N, E]
    }
}

// ---------------- K2: bilinear 24x24 -> 64x64 (half-pixel, edge clamp)
// src grid[c][sy][sx] = x[(sy*24+sx)*64 + c]; dst g: [64ch][64][64]
__global__ void k_resize(const float* __restrict__ x, float* __restrict__ g) {
    int idx = blockIdx.x * blockDim.x + threadIdx.x;
    if (idx >= EE * GG * GG) return;
    int c  = idx >> 12;
    int gy = (idx >> 6) & 63;
    int gx = idx & 63;
    const float scale = (float)SIDE / (float)GG;   // 0.375
    float sy = (gy + 0.5f) * scale - 0.5f;
    float sx = (gx + 0.5f) * scale - 0.5f;
    int y0 = (int)floorf(sy); float fy = sy - (float)y0;
    int x0 = (int)floorf(sx); float fx = sx - (float)x0;
    int y1 = y0 + 1, x1 = x0 + 1;
    y0 = min(max(y0, 0), SIDE - 1); y1 = min(max(y1, 0), SIDE - 1);
    x0 = min(max(x0, 0), SIDE - 1); x1 = min(max(x1, 0), SIDE - 1);
    float v00 = x[(y0 * SIDE + x0) * EE + c];
    float v01 = x[(y0 * SIDE + x1) * EE + c];
    float v10 = x[(y1 * SIDE + x0) * EE + c];
    float v11 = x[(y1 * SIDE + x1) * EE + c];
    g[idx] = (1.f - fy) * ((1.f - fx) * v00 + fx * v01)
           +        fy  * ((1.f - fx) * v10 + fx * v11);
}

// ---------------- K3/K4: direct 3x3 SAME conv, 64ch -> 64ch on 64x64
__global__ void k_conv3x3(const float* __restrict__ in, const float* __restrict__ w,
                          const float* __restrict__ bias, float* __restrict__ out,
                          int do_relu) {
    int idx = blockIdx.x * blockDim.x + threadIdx.x;
    if (idx >= EE * GG * GG) return;
    int o = idx >> 12;
    int y = (idx >> 6) & 63;
    int x = idx & 63;
    float acc = bias[o];
    const float* wbase = w + o * EE * 9;
    for (int i = 0; i < EE; ++i) {
        const float* ip = in + i * (GG * GG);
        const float* wr = wbase + i * 9;
        #pragma unroll
        for (int ky = 0; ky < 3; ++ky) {
            int yy = y + ky - 1;
            if (yy < 0 || yy > 63) continue;
            #pragma unroll
            for (int kx = 0; kx < 3; ++kx) {
                int xx = x + kx - 1;
                if (xx < 0 || xx > 63) continue;
                acc += wr[ky * 3 + kx] * ip[yy * 64 + xx];
            }
        }
    }
    if (do_relu) acc = fmaxf(acc, 0.f);
    out[idx] = acc;
}

// ---------------- K5: per-neuron refinement + projections -> A, Bt, Aw, Bw
__global__ void k_feats(const float* __restrict__ z2,
                        const float* __restrict__ rw1, const float* __restrict__ rb1,
                        const float* __restrict__ rw2, const float* __restrict__ rb2,
                        const float* __restrict__ spw, const float* __restrict__ spb,
                        const float* __restrict__ tpw, const float* __restrict__ tpb,
                        const float* __restrict__ ew1, const float* __restrict__ eb1,
                        const float* __restrict__ wmw1, const float* __restrict__ wmb1,
                        float* __restrict__ A, float* __restrict__ Bt,
                        float* __restrict__ Aw, float* __restrict__ Bw) {
    __shared__ float f[EE], h1[EE], emb[EE], s[HH], tg[HH];
    int n = blockIdx.x;     // 576
    int t = threadIdx.x;    // 256
    if (t < EE) f[t] = z2[t * (GG * GG) + n];   // feats[n,c] = z2[c][n]
    __syncthreads();
    if (t < EE) {
        float a = rb1[t];
        const float* wr = rw1 + t * EE;
        for (int k = 0; k < EE; ++k) a += f[k] * wr[k];
        h1[t] = fmaxf(a, 0.f);
    }
    __syncthreads();
    if (t < EE) {
        float a = rb2[t];
        const float* wr = rw2 + t * EE;
        for (int k = 0; k < EE; ++k) a += h1[k] * wr[k];
        emb[t] = a;
    }
    __syncthreads();
    {
        float a = spb[t], c = tpb[t];
        const float* ws = spw + t * EE;
        const float* wt = tpw + t * EE;
        for (int k = 0; k < EE; ++k) { a += emb[k] * ws[k]; c += emb[k] * wt[k]; }
        s[t] = a; tg[t] = c;
    }
    __syncthreads();
    // row n, col t of the four broadcast operands
    float a = 0.f, bt = eb1[t], aw = 0.f, bw = wmb1[t];
    const float* e1  = ew1  + t * (2 * HH);
    const float* w1r = wmw1 + t * (2 * HH);
    for (int k = 0; k < HH; ++k) {
        a  += s[k]  * e1[k];
        bt += tg[k] * e1[HH + k];
        aw += s[k]  * w1r[k];
        bw += tg[k] * w1r[HH + k];
    }
    A [n * HH + t] = a;
    Bt[n * HH + t] = bt;
    Aw[n * HH + t] = aw;
    Bw[n * HH + t] = bw;
}

// ---------------- K6: pairwise edge MLP via WMMA f16 (the 21.7 GFLOP stage)
// grid: (36 j-tiles, 144 i-groups), block: 128 thr = 4 waves; wave w owns i0+w.
__global__ void __launch_bounds__(128)
k_edge(const float* __restrict__ A,  const float* __restrict__ Bt,
       const float* __restrict__ Aw, const float* __restrict__ Bw,
       const _Float16* __restrict__ w2t,            // [256 K][128 N] f16
       const float* __restrict__ eb2, const float* __restrict__ ew3,
       const float* __restrict__ eb3, const float* __restrict__ wmw2,
       const float* __restrict__ wmb2,
       float* __restrict__ adj, float* __restrict__ wout) {
    __shared__ float sBt[16][HH];
    __shared__ float sBw[16][HH];
    __shared__ float sA [4][HH];
    __shared__ float sAw[4][HH];
    __shared__ float sW3[128];
    __shared__ float sB2[128];
    __shared__ float sWm[HH];

    int j0 = blockIdx.x * 16;
    int i0 = blockIdx.y * 4;
    int t  = threadIdx.x;

    for (int k = t; k < 16 * HH; k += 128) {
        int row = k >> 8, col = k & 255;
        sBt[row][col] = Bt[(j0 + row) * HH + col];
        sBw[row][col] = Bw[(j0 + row) * HH + col];
    }
    for (int k = t; k < 4 * HH; k += 128) {
        int row = k >> 8, col = k & 255;
        sA [row][col] = A [(i0 + row) * HH + col];
        sAw[row][col] = Aw[(i0 + row) * HH + col];
    }
    sW3[t] = ew3[t];
    sB2[t] = eb2[t];
    for (int k = t; k < HH; k += 128) sWm[k] = wmw2[k];
    __syncthreads();

    int wave = t >> 5;            // 0..3 -> i = i0 + wave
    int lane = t & 31;
    int m    = lane & 15;         // pair row within tile (A/M mapping)
    int kh   = lane >> 4;         // K-half for the A fragment
    int i    = i0 + wave;

    const float* Ai  = sA[wave];
    const float* Btm = sBt[m];

    v8f acc[8];
    v8f zero = {0.f, 0.f, 0.f, 0.f, 0.f, 0.f, 0.f, 0.f};
    #pragma unroll
    for (int nt = 0; nt < 8; ++nt) acc[nt] = zero;

    for (int kc = 0; kc < 8; ++kc) {            // K = 256 in chunks of 32
        // A fragment: lane = {M=m, Khalf=kh}; elements 0..7 -> k_local 8kh..8kh+7,
        // elements 8..15 -> k_local 16+8kh..23+8kh (ISA 16-bit A 16x32 layout)
        int kb = kc * 32 + kh * 8;
        v16h af;
        #pragma unroll
        for (int e = 0; e < 8; ++e) {
            float v0 = fmaxf(Ai[kb + e]      + Btm[kb + e],      0.f);
            float v1 = fmaxf(Ai[kb + 16 + e] + Btm[kb + 16 + e], 0.f);
            af[e]     = (_Float16)v0;
            af[8 + e] = (_Float16)v1;
        }
        // B fragment: lane = {N=m, Khalf=kh}; elements 0..7 -> k_local 8kh..8kh+7,
        // elements 8..15 -> k_local 16+8kh.. (same K order as A fragment).
        const _Float16* bcol = w2t + (kc * 32 + kh * 8) * 128 + m;
        #pragma unroll
        for (int nt = 0; nt < 8; ++nt) {
            v16h bf;
            #pragma unroll
            for (int e = 0; e < 8; ++e) {
                bf[e]     = bcol[e * 128 + nt * 16];
                bf[8 + e] = bcol[(16 + e) * 128 + nt * 16];
            }
            acc[nt] = __builtin_amdgcn_wmma_f32_16x16x32_f16(
                false, af, false, bf, (short)0, acc[nt], false, false);
        }
    }

    // Epilogue: he2 = relu(acc + b2); partial logit = he2 . w3 (N on lane&15)
    float sred[8];
    #pragma unroll
    for (int r = 0; r < 8; ++r) sred[r] = 0.f;
    #pragma unroll
    for (int nt = 0; nt < 8; ++nt) {
        int ncol = nt * 16 + m;
        float b2v = sB2[ncol];
        float w3v = sW3[ncol];
        #pragma unroll
        for (int r = 0; r < 8; ++r) {
            float h2 = fmaxf(acc[nt][r] + b2v, 0.f);
            sred[r] += h2 * w3v;
        }
    }
    // reduce the 16 N-lanes within each half-wave (C/D layout: M = r + 8*kh)
    #pragma unroll
    for (int r = 0; r < 8; ++r) {
        float v = sred[r];
        for (int off = 1; off < 16; off <<= 1) v += __shfl_xor(v, off, 16);
        sred[r] = v;
    }
    if (m == 0) {
        float b3 = eb3[0];
        #pragma unroll
        for (int r = 0; r < 8; ++r) {
            int jj = j0 + r + 8 * kh;
            float lg = sred[r] + b3;
            adj[i * NN + jj] = 1.f / (1.f + expf(-lg));
        }
    }

    // Weight MLP (cheap): lane < 16 handles pair (i, j0+lane)
    if (lane < 16) {
        const float* Awi = sAw[wave];
        const float* Bwm = sBw[lane];
        float a = 0.f;
        for (int k = 0; k < HH; ++k)
            a += fmaxf(Awi[k] + Bwm[k], 0.f) * sWm[k];
        a += wmb2[0];
        // stable softplus = max(x,0) + log1p(exp(-|x|))
        float sp = fmaxf(a, 0.f) + log1pf(expf(-fabsf(a)));
        wout[i * NN + (j0 + lane)] = sp;
    }
}

// ---------------------------------------------------------------------------
extern "C" void kernel_launch(void* const* d_in, const int* in_sizes, int n_in,
                              void* d_out, int out_size, void* d_ws, size_t ws_size,
                              hipStream_t stream) {
    (void)in_sizes; (void)n_in; (void)out_size; (void)ws_size;

    const float* raster = (const float*)d_in[0];
    const float* ip_w1  = (const float*)d_in[1];
    const float* ip_b1  = (const float*)d_in[2];
    const float* ip_w2  = (const float*)d_in[3];
    const float* ip_b2  = (const float*)d_in[4];
    const float* c1_w   = (const float*)d_in[5];
    const float* c1_b   = (const float*)d_in[6];
    const float* c2_w   = (const float*)d_in[7];
    const float* c2_b   = (const float*)d_in[8];
    const float* ref_w1 = (const float*)d_in[9];
    const float* ref_b1 = (const float*)d_in[10];
    const float* ref_w2 = (const float*)d_in[11];
    const float* ref_b2 = (const float*)d_in[12];
    const float* sp_w   = (const float*)d_in[13];
    const float* sp_b   = (const float*)d_in[14];
    const float* tp_w   = (const float*)d_in[15];
    const float* tp_b   = (const float*)d_in[16];
    const float* e_w1   = (const float*)d_in[17];
    const float* e_b1   = (const float*)d_in[18];
    const float* e_w2   = (const float*)d_in[19];
    const float* e_b2   = (const float*)d_in[20];
    const float* e_w3   = (const float*)d_in[21];
    const float* e_b3   = (const float*)d_in[22];
    const float* wm_w1  = (const float*)d_in[23];
    const float* wm_b1  = (const float*)d_in[24];
    const float* wm_w2  = (const float*)d_in[25];
    const float* wm_b2  = (const float*)d_in[26];

    // workspace layout (floats)
    float* ws    = (float*)d_ws;
    float* x     = ws;                         // 576*64      = 36864
    float* gridr = x     + NN * EE;            // 64*64*64    = 262144
    float* z1    = gridr + EE * GG * GG;       // 262144
    float* z2    = z1    + EE * GG * GG;       // 262144
    float* Abuf  = z2    + EE * GG * GG;       // 576*256     = 147456
    float* Btbuf = Abuf  + NN * HH;
    float* Awbuf = Btbuf + NN * HH;
    float* Bwbuf = Awbuf + NN * HH;
    _Float16* w2t = (_Float16*)(Bwbuf + NN * HH);   // 256*128 f16 (32B aligned)

    float* adj = (float*)d_out;                // [576,576]
    float* wts = adj + NN * NN;                // [576,576]

    k_cvt_w2<<<(128 * 256 + 255) / 256, 256, 0, stream>>>(e_w2, w2t);
    k_inproj<<<NN, 128, 0, stream>>>(raster, ip_w1, ip_b1, ip_w2, ip_b2, x);
    k_resize<<<(EE * GG * GG + 255) / 256, 256, 0, stream>>>(x, gridr);
    k_conv3x3<<<(EE * GG * GG + 255) / 256, 256, 0, stream>>>(gridr, c1_w, c1_b, z1, 1);
    k_conv3x3<<<(EE * GG * GG + 255) / 256, 256, 0, stream>>>(z1, c2_w, c2_b, z2, 0);
    k_feats<<<NN, 256, 0, stream>>>(z2, ref_w1, ref_b1, ref_w2, ref_b2,
                                    sp_w, sp_b, tp_w, tp_b,
                                    e_w1, e_b1, wm_w1, wm_b1,
                                    Abuf, Btbuf, Awbuf, Bwbuf);
    k_edge<<<dim3(NN / 16, NN / 4), 128, 0, stream>>>(Abuf, Btbuf, Awbuf, Bwbuf,
                                                      w2t, e_b2, e_w3, e_b3,
                                                      wm_w2, wm_b2, adj, wts);
}
